// Head_23871428231855
// MI455X (gfx1250) — compile-verified, matrix operand order checked by reading, not run on hardware
//
#include <hip/hip_runtime.h>

typedef __attribute__((ext_vector_type(16))) _Float16 v16h;
typedef __attribute__((ext_vector_type(8)))  _Float16 v8h;
typedef __attribute__((ext_vector_type(8)))  float    v8f;
typedef __attribute__((ext_vector_type(4)))  float    v4f;

#define B_SZ 4
#define T_SZ 2048
#define E_SZ 1024
#define D_SZ 64
#define M_TOT (B_SZ * T_SZ)   /* 8192 token rows */

// ---- CDNA5 TDM availability probes (safe: fallback path always compiles) ----
#if __has_builtin(__builtin_amdgcn_tensor_load_to_lds)
#define HAVE_TDM 1
#if __has_include(<hip/amd_detail/amd_gfx1250_TDM.h>)
#define TDM_6ARG 1   /* amdgpu-toolchain / therock-10.0 flavor */
#endif
#endif

#if __has_builtin(__builtin_amdgcn_s_wait_tensorcnt)
#define WAIT_TENSORCNT(n) __builtin_amdgcn_s_wait_tensorcnt((short)(n))
#else
#define WAIT_TENSORCNT(n) asm volatile("s_wait_tensorcnt %0" :: "i"(n) : "memory")
#endif

// Build a 16-half A/B operand from two contiguous 16-byte runs
// (CDNA5 16-bit operand layout: VGPRs 0-3 hold K=base..base+7,
//  VGPRs 4-7 hold K=base+16..base+23, base = 8*(lane>>4)).
static __device__ __forceinline__ v16h ld16(const _Float16* lo, const _Float16* hi) {
  v8h a = *(const v8h*)lo;
  v8h b = *(const v8h*)hi;
  return __builtin_shufflevector(a, b, 0,1,2,3,4,5,6,7,8,9,10,11,12,13,14,15);
}

#ifdef HAVE_TDM
typedef __attribute__((ext_vector_type(4))) unsigned int v4u;
typedef __attribute__((ext_vector_type(8))) int          v8i;
typedef __attribute__((ext_vector_type(4))) int          v4i;

// Issue one 2-D TDM tile load (f16 elements) into LDS.
// D# per CDNA5 ISA 8.3/8.4: group0 = {count=1 | lds_addr | global_addr | type=2},
// group1 = {data_size=2B, tensor dims, tile dims, dim0 stride}; groups 2/3 zero (<=2D).
static __device__ __forceinline__ void tdm_load_2d(
    unsigned lds_off, const void* gptr,
    unsigned tile_d0, unsigned tile_d1,
    unsigned tensor_d0, unsigned tensor_d1, unsigned stride0)
{
  unsigned long long ga = (unsigned long long)gptr;
  v4u g0;
  g0[0] = 1u;                                             // count=1 (valid user D#)
  g0[1] = lds_off;                                        // LDS byte address
  g0[2] = (unsigned)(ga & 0xFFFFFFFFu);                   // global_addr[31:0]
  g0[3] = (unsigned)((ga >> 32) & 0x01FFFFFFu) | (2u << 30); // addr[56:32] | type=2
  v8i g1;
  g1[0] = (int)(1u << 16);                                // data_size code 1 = 2 bytes
  g1[1] = (int)(tensor_d0 << 16);                         // tensor_dim0[15:0]
  g1[2] = (int)((tensor_d0 >> 16) | (tensor_d1 << 16));   // dim0[31:16] | dim1[15:0]
  g1[3] = (int)((tensor_d1 >> 16) | (tile_d0 << 16));     // dim1[31:16] | tile_dim0
  g1[4] = (int)(tile_d1 & 0xFFFFu);                       // tile_dim1 (tile_dim2 = 0)
  g1[5] = (int)stride0;                                   // tensor_dim0_stride[31:0]
  g1[6] = 0;
  g1[7] = 0;
  v4i gz = {};
#ifdef TDM_6ARG
  v8i gz8 = {};
  __builtin_amdgcn_tensor_load_to_lds(g0, g1, gz, gz, gz8, 0);
#else
  __builtin_amdgcn_tensor_load_to_lds(g0, g1, gz, gz, 0);
#endif
}
#endif // HAVE_TDM

// ---------------- weight transpose+convert (tiny, runs once) ----------------
// wt[w*65536 + d*1024 + k] = (f16) W_w[k*64 + d]
__global__ void cvt_w_kernel(const float* __restrict__ Wk, const float* __restrict__ Wq,
                             const float* __restrict__ Wv, _Float16* __restrict__ wt) {
  int i = blockIdx.x * blockDim.x + threadIdx.x;
  if (i >= 3 * E_SZ * D_SZ) return;
  int w   = i >> 16;
  int rem = i & 65535;
  int d   = rem >> 10;
  int k   = rem & 1023;
  const float* W = (w == 0) ? Wk : (w == 1) ? Wq : Wv;
  wt[i] = (_Float16)W[k * D_SZ + d];
}

// ---------------- QKV projection: one wave -> one 16-row tile ----------------
// Reads x directly in f32, converts to f16 in-register (v_cvt_pk_rtz), and
// software-pipelines the B tiles with an explicit 2-chunk ping-pong (6 B-groups,
// even count) so the double-buffer hand-off is pure register renaming --
// no v_mov copies, no WMMA->VALU WAR hazard NOPs in the hot loop.

__global__ __launch_bounds__(32) void qkv_kernel(
    const float* __restrict__ x, const _Float16* __restrict__ wt,
    const float* __restrict__ bk, const float* __restrict__ bq, const float* __restrict__ bv,
    _Float16* __restrict__ qh, _Float16* __restrict__ kh, _Float16* __restrict__ vt)
{
  const int mtile = blockIdx.x;          // 512 tiles of 16 rows
  const int lane  = threadIdx.x;
  const int half  = lane & 15;
  const int grp   = lane >> 4;
  const int kg    = grp * 8;

  const float* xrow = x + (size_t)(mtile * 16 + half) * E_SZ;

  v8f zero = {};
  v8f acc[3][4];
  #pragma unroll
  for (int w = 0; w < 3; ++w)
    #pragma unroll
    for (int nt = 0; nt < 4; ++nt) acc[w][nt] = zero;

  // A chunk: 16 f32 -> 16 f16 (A-operand layout), via packed RTZ converts
  auto ldA = [&](int kc) -> v16h {
    const float* p0 = xrow + kc + kg;
    const float* p1 = xrow + kc + 16 + kg;
    v4f f0 = *(const v4f*)p0, f1 = *(const v4f*)(p0 + 4);
    v4f f2 = *(const v4f*)p1, f3 = *(const v4f*)(p1 + 4);
    v16h r;
    { auto c = __builtin_amdgcn_cvt_pkrtz(f0.x, f0.y); r[0]  = c[0]; r[1]  = c[1]; }
    { auto c = __builtin_amdgcn_cvt_pkrtz(f0.z, f0.w); r[2]  = c[0]; r[3]  = c[1]; }
    { auto c = __builtin_amdgcn_cvt_pkrtz(f1.x, f1.y); r[4]  = c[0]; r[5]  = c[1]; }
    { auto c = __builtin_amdgcn_cvt_pkrtz(f1.z, f1.w); r[6]  = c[0]; r[7]  = c[1]; }
    { auto c = __builtin_amdgcn_cvt_pkrtz(f2.x, f2.y); r[8]  = c[0]; r[9]  = c[1]; }
    { auto c = __builtin_amdgcn_cvt_pkrtz(f2.z, f2.w); r[10] = c[0]; r[11] = c[1]; }
    { auto c = __builtin_amdgcn_cvt_pkrtz(f3.x, f3.y); r[12] = c[0]; r[13] = c[1]; }
    { auto c = __builtin_amdgcn_cvt_pkrtz(f3.z, f3.w); r[14] = c[0]; r[15] = c[1]; }
    return r;
  };

  auto ldB = [&](int w, int kc, v16h bg[4]) {
    #pragma unroll
    for (int nt = 0; nt < 4; ++nt) {
      const _Float16* wrow =
          wt + (size_t)w * (E_SZ * D_SZ) + (size_t)(nt * 16 + half) * E_SZ + kc;
      bg[nt] = ld16(wrow + kg, wrow + 16 + kg);
    }
  };

  auto mm4 = [&](int w, const v16h& a, v16h bg[4]) {
    #pragma unroll
    for (int nt = 0; nt < 4; ++nt)
      acc[w][nt] = __builtin_amdgcn_wmma_f32_16x16x32_f16(
          false, a, false, bg[nt], (short)0, acc[w][nt], false, false);
  };

  v16h a0 = ldA(0);
  v16h b0[4], b1[4];
  ldB(0, 0, b0);

  // 32 chunks of K=32, processed as 16 pairs; kc+32 is always in range.
  for (int kc = 0; kc < E_SZ; kc += 64) {
    v16h a1 = ldA(kc + 32);                // A for odd chunk, loaded early

    ldB(1, kc, b1);       mm4(0, a0, b0);  // (0,kc):      consume b0, stage (1,kc)
    ldB(2, kc, b0);       mm4(1, a0, b1);  // (1,kc):      consume b1, stage (2,kc)
    ldB(0, kc + 32, b1);  mm4(2, a0, b0);  // (2,kc):      consume b0, stage (0,kc+32)
    ldB(1, kc + 32, b0);  mm4(0, a1, b1);  // (0,kc+32):   consume b1, stage (1,kc+32)
    ldB(2, kc + 32, b1);  mm4(1, a1, b0);  // (1,kc+32):   consume b0, stage (2,kc+32)
    if (kc + 64 < E_SZ) { ldB(0, kc + 64, b0); }
    mm4(2, a1, b1);                        // (2,kc+32):   consume b1
    if (kc + 64 < E_SZ) { a0 = ldA(kc + 64); }
  }

  #pragma unroll
  for (int w = 0; w < 3; ++w) {
    #pragma unroll
    for (int nt = 0; nt < 4; ++nt) {
      const int n = nt * 16 + half;
      const float bias = (w == 0 ? bk : w == 1 ? bq : bv)[n];
      #pragma unroll
      for (int r = 0; r < 8; ++r) {
        const int m = mtile * 16 + r + 8 * grp;        // C-layout row
        const _Float16 val = (_Float16)(acc[w][nt][r] + bias);
        if (w == 0)      kh[(size_t)m * D_SZ + n] = val;
        else if (w == 1) qh[(size_t)m * D_SZ + n] = val;
        else {                                         // V stored transposed per batch
          const int bb = m >> 11, tt = m & (T_SZ - 1);
          vt[((size_t)bb * D_SZ + n) * T_SZ + tt] = val;
        }
      }
    }
  }
}

// ---------------- flash attention: one wave -> 16 query rows ----------------
// TDM path: double-buffered tensor_load_to_lds stages the 32x64 K tile and the
// 64x32 V^T tile (4 KB each) one key-block ahead; s_wait_tensorcnt gates reads.

#define SM_PLDS  0
#define SM_K0    1024
#define SM_K1    (1024 + 4096)
#define SM_V0    (1024 + 2 * 4096)
#define SM_V1    (1024 + 3 * 4096)
#define SM_TOTAL (1024 + 4 * 4096)

__global__ __launch_bounds__(32) void attn_kernel(
    const _Float16* __restrict__ qh, const _Float16* __restrict__ kh,
    const _Float16* __restrict__ vt, float* __restrict__ out)
{
  __shared__ __attribute__((aligned(16))) char smem[SM_TOTAL];
  _Float16* plds = (_Float16*)(smem + SM_PLDS);

  const int tile = blockIdx.x;            // 512 = 4 batches * 128 query tiles
  const int b    = tile >> 7;
  const int qb   = (tile & 127) * 16;
  const int lane = threadIdx.x;
  const int half = lane & 15;
  const int grp  = lane >> 4;
  const int kg   = grp * 8;

  // Q operand, A-layout, K = D = 64 (two 16x32 chunks), resident all loop long
  const _Float16* qrow = qh + (size_t)(b * T_SZ + qb + half) * D_SZ;
  const v16h qa0 = ld16(qrow + kg,      qrow + 16 + kg);
  const v16h qa1 = ld16(qrow + 32 + kg, qrow + 48 + kg);

  v8f zero = {};
  float mrow[8], lrow[8];
  v8f oacc[4];
  #pragma unroll
  for (int r = 0; r < 8; ++r) { mrow[r] = -1e30f; lrow[r] = 0.0f; }
  #pragma unroll
  for (int nt = 0; nt < 4; ++nt) oacc[nt] = zero;

  const int tqmax = qb + 15;

#ifdef HAVE_TDM
  // preload key-block 0 into buffer 0
  tdm_load_2d(SM_K0, kh + (size_t)(b * T_SZ) * D_SZ, 64, 32, 64, 32, 64);
  tdm_load_2d(SM_V0, vt + (size_t)(b * D_SZ) * T_SZ, 32, 64, 32, 64, T_SZ);
#endif

  for (int sb = 0; sb <= tqmax; sb += 32) {        // causal: skip blocks above diagonal
    const int buf = (sb >> 5) & 1;

#ifdef HAVE_TDM
    const int nsb = sb + 32;
    if (nsb <= tqmax) {                            // stage next block, then drain current
      tdm_load_2d(buf ? SM_K0 : SM_K1, kh + (size_t)(b * T_SZ + nsb) * D_SZ,
                  64, 32, 64, 32, 64);
      tdm_load_2d(buf ? SM_V0 : SM_V1, vt + (size_t)(b * D_SZ) * T_SZ + nsb,
                  32, 64, 32, 64, T_SZ);
      WAIT_TENSORCNT(2);
    } else {
      WAIT_TENSORCNT(0);
    }
    asm volatile("" ::: "memory");
    const _Float16* kbase = (const _Float16*)(smem + (buf ? SM_K1 : SM_K0));
    const _Float16* vbase = (const _Float16*)(smem + (buf ? SM_V1 : SM_V0));
    const int kstride = 64, vstride = 32;
#else
    const _Float16* kbase = kh + (size_t)(b * T_SZ + sb) * D_SZ;
    const _Float16* vbase = vt + (size_t)(b * D_SZ) * T_SZ + sb;
    const int kstride = 64, vstride = T_SZ;
    if (sb + 32 <= tqmax) {                        // gfx1250 global_prefetch_b8
      __builtin_prefetch(kbase + 32 * 64, 0, 0);
      __builtin_prefetch(vbase + 32, 0, 0);
    }
#endif

    // ---- scores = q . k^T : two 16x16 tiles over K=64 ----
    v8f st[2];
    #pragma unroll
    for (int c = 0; c < 2; ++c) {
      const _Float16* krow = kbase + (size_t)(c * 16 + half) * kstride;
      const v16h kb0 = ld16(krow + kg,      krow + 16 + kg);
      const v16h kb1 = ld16(krow + 32 + kg, krow + 48 + kg);
      v8f s = zero;
      s = __builtin_amdgcn_wmma_f32_16x16x32_f16(false, qa0, false, kb0, (short)0, s, false, false);
      s = __builtin_amdgcn_wmma_f32_16x16x32_f16(false, qa1, false, kb1, (short)0, s, false, false);
      st[c] = s;
    }

    // ---- scale 1/sqrt(64) + causal mask ----
    #pragma unroll
    for (int c = 0; c < 2; ++c) {
      const int skey = sb + c * 16 + half;
      #pragma unroll
      for (int r = 0; r < 8; ++r) {
        const int tq = qb + r + 8 * grp;
        const float v = st[c][r] * 0.125f;
        st[c][r] = (skey > tq) ? -1e30f : v;
      }
    }

    // ---- online softmax (row reduce = 16-lane butterfly) ----
    float bm[8];
    #pragma unroll
    for (int r = 0; r < 8; ++r) bm[r] = fmaxf(st[0][r], st[1][r]);
    #pragma unroll
    for (int off = 8; off >= 1; off >>= 1)
      #pragma unroll
      for (int r = 0; r < 8; ++r)
        bm[r] = fmaxf(bm[r], __shfl_xor(bm[r], off, 16));

    float p0[8], p1[8], corr[8];
    #pragma unroll
    for (int r = 0; r < 8; ++r) {
      const float mn = fmaxf(mrow[r], bm[r]);
      corr[r] = __expf(mrow[r] - mn);
      mrow[r] = mn;
      p0[r] = __expf(st[0][r] - mn);
      p1[r] = __expf(st[1][r] - mn);
    }
    float bs[8];
    #pragma unroll
    for (int r = 0; r < 8; ++r) bs[r] = p0[r] + p1[r];
    #pragma unroll
    for (int off = 8; off >= 1; off >>= 1)
      #pragma unroll
      for (int r = 0; r < 8; ++r)
        bs[r] += __shfl_xor(bs[r], off, 16);
    #pragma unroll
    for (int r = 0; r < 8; ++r) lrow[r] = lrow[r] * corr[r] + bs[r];
    #pragma unroll
    for (int nt = 0; nt < 4; ++nt)
      #pragma unroll
      for (int r = 0; r < 8; ++r) oacc[nt][r] *= corr[r];

    // ---- C-layout P -> A-layout P through LDS (16x32 f16 = 1 KB) ----
    #pragma unroll
    for (int r = 0; r < 8; ++r) {
      plds[(r + 8 * grp) * 32 + half]      = (_Float16)p0[r];
      plds[(r + 8 * grp) * 32 + 16 + half] = (_Float16)p1[r];
    }
    __syncthreads();
    const v16h pa = ld16(&plds[half * 32 + kg], &plds[half * 32 + 16 + kg]);
    __syncthreads();

    // ---- O += P . V  (V^T: contiguous along key index) ----
    #pragma unroll
    for (int nt = 0; nt < 4; ++nt) {
      const _Float16* vrow = vbase + (size_t)(nt * 16 + half) * vstride;
      const v16h vb = ld16(vrow + kg, vrow + 16 + kg);
      oacc[nt] = __builtin_amdgcn_wmma_f32_16x16x32_f16(
          false, pa, false, vb, (short)0, oacc[nt], false, false);
    }
  }

  // ---- normalize and write fp32 output ----
  #pragma unroll
  for (int nt = 0; nt < 4; ++nt) {
    const int d = nt * 16 + half;
    #pragma unroll
    for (int r = 0; r < 8; ++r) {
      const int tq = qb + r + 8 * grp;
      out[((size_t)(b * T_SZ + tq)) * D_SZ + d] = oacc[nt][r] / lrow[r];
    }
  }
}

// ---------------- host-side launcher ----------------

extern "C" void kernel_launch(void* const* d_in, const int* in_sizes, int n_in,
                              void* d_out, int out_size, void* d_ws, size_t ws_size,
                              hipStream_t stream) {
  const float* x  = (const float*)d_in[0];
  const float* Wk = (const float*)d_in[1];
  const float* bk = (const float*)d_in[2];
  const float* Wq = (const float*)d_in[3];
  const float* bq = (const float*)d_in[4];
  const float* Wv = (const float*)d_in[5];
  const float* bv = (const float*)d_in[6];
  float* out = (float*)d_out;

  // workspace (f16): wt 384 KiB (+pad to 512 KiB) | qh 1 MiB | kh 1 MiB | vt 1 MiB
  char* ws = (char*)d_ws;
  _Float16* wt = (_Float16*)ws;
  _Float16* qh = (_Float16*)(ws + 512 * 1024);
  _Float16* kh = qh + (size_t)M_TOT * D_SZ;
  _Float16* vt = kh + (size_t)M_TOT * D_SZ;

  cvt_w_kernel<<<(3 * E_SZ * D_SZ + 255) / 256, 256, 0, stream>>>(Wk, Wq, Wv, wt);
  qkv_kernel<<<M_TOT / 16, 32, 0, stream>>>(x, wt, bk, bq, bv, qh, kh, vt);
  attn_kernel<<<M_TOT / 16, 32, 0, stream>>>(qh, kh, vt, out);
}
